// MixtureOfExperts_47699906789405
// MI455X (gfx1250) — compile-verified, matrix operand order checked by reading, not run on hardware
//
#include <hip/hip_runtime.h>
#include <cmath>

// ---------------- problem constants (from reference setup) ----------------
namespace {
constexpr int Tt   = 4 * 2048;     // tokens = B*S = 8192
constexpr int D    = 1024;
constexpr int E    = 8;
constexpr int F    = 4096;
constexpr int TOPK = 2;
constexpr int TSEL = Tt * TOPK;    // 16384 routed rows
constexpr int TM = 128, TN = 128, KB = 64;
constexpr int MAX_TILES = TSEL / TM;   // 128 (worst case: all rows -> one expert)
}

typedef __bf16 bf16;
typedef bf16  v16bf __attribute__((ext_vector_type(16)));
typedef bf16  v8bf  __attribute__((ext_vector_type(8)));
typedef float v8f   __attribute__((ext_vector_type(8)));

union FragBF { v16bf v; v8bf h[2]; };

// -------- CDNA5 async copy: global -> LDS, tracked by ASYNCcnt --------
__device__ __forceinline__ unsigned ldsoff_of(const void* p) {
    // flat shared pointer = {SHARED_BASE, lds_offset}; low 32 bits are the DS offset
    return (unsigned)(size_t)p;
}
__device__ __forceinline__ void async_b128(unsigned ldsoff, const void* g) {
    asm volatile("global_load_async_to_lds_b128 %0, %1, off"
                 :: "v"(ldsoff), "v"((unsigned long long)(size_t)g)
                 : "memory");
}
__device__ __forceinline__ void async_wait0() {
    asm volatile("s_wait_asynccnt 0" ::: "memory");
}

// ---------------- init: zero counts & cursors ----------------
__global__ void moe_init(int* counts, int* cursor) {
    int i = threadIdx.x;
    if (i < E) { counts[i] = 0; cursor[i] = 0; }
}

// ---------------- f32 -> bf16 streaming convert (x) ----------------
__global__ void moe_cvt_bf16(const float* __restrict__ src, bf16* __restrict__ dst, int n) {
    for (int i = blockIdx.x * blockDim.x + threadIdx.x; i < n; i += gridDim.x * blockDim.x)
        dst[i] = (bf16)src[i];
}

// ------- f32 [E][R][C] -> bf16 [E][C][R] transpose-convert (weights) -------
// 32x32 tile via LDS; 256 threads (32x8). Makes GEMM B-tiles K-contiguous so
// staging is a pure row copy (async-LDS friendly, no in-loop transpose).
__global__ void moe_cvt_tr(const float* __restrict__ src, bf16* __restrict__ dst,
                           int R, int C) {
    __shared__ float tile[32][33];
    const int e  = blockIdx.z;
    const int r0 = blockIdx.y * 32;
    const int c0 = blockIdx.x * 32;
    const int tx = threadIdx.x & 31;
    const int ty = threadIdx.x >> 5;
    const float* s = src + (size_t)e * R * C;
    bf16*       d  = dst + (size_t)e * R * C;
#pragma unroll
    for (int i = ty; i < 32; i += 8)
        tile[i][tx] = s[(size_t)(r0 + i) * C + c0 + tx];
    __syncthreads();
#pragma unroll
    for (int i = ty; i < 32; i += 8)
        d[(size_t)(c0 + i) * R + r0 + tx] = (bf16)tile[tx][i];
}

// ---------------- router: wave per token, top-2 + softmax ----------------
__global__ void moe_router(const float* __restrict__ x, const float* __restrict__ gw,
                           float* __restrict__ top_w, int* __restrict__ top_e,
                           int* __restrict__ counts) {
    int lane = threadIdx.x & 31;
    int wid  = threadIdx.x >> 5;
    int t = blockIdx.x * (blockDim.x >> 5) + wid;   // one wave per token
    if (t >= Tt) return;

    float acc[E];
#pragma unroll
    for (int e = 0; e < E; ++e) acc[e] = 0.f;
    for (int d = lane; d < D; d += 32) {
        float xv = x[(size_t)t * D + d];
#pragma unroll
        for (int e = 0; e < E; ++e) acc[e] += xv * gw[d * E + e];
    }
#pragma unroll
    for (int off = 16; off > 0; off >>= 1) {
#pragma unroll
        for (int e = 0; e < E; ++e) acc[e] += __shfl_xor(acc[e], off, 32);
    }
    if (lane == 0) {
        float v0 = -INFINITY, v1 = -INFINITY; int i0 = 0, i1 = 0;
#pragma unroll
        for (int e = 0; e < E; ++e) {
            float v = acc[e];
            if (v > v0)      { v1 = v0; i1 = i0; v0 = v; i0 = e; }
            else if (v > v1) { v1 = v;  i1 = e; }
        }
        float w0 = 1.f / (1.f + __expf(v1 - v0));   // softmax over top-2, v0 >= v1
        top_e[t * 2 + 0] = i0; top_w[t * 2 + 0] = w0;
        top_e[t * 2 + 1] = i1; top_w[t * 2 + 1] = 1.f - w0;
        atomicAdd(&counts[i0], 1);
        atomicAdd(&counts[i1], 1);
    }
}

// ---------------- exclusive scan over E=8 (tiny) ----------------
__global__ void moe_offsets(const int* __restrict__ counts, int* __restrict__ offs) {
    if (threadIdx.x == 0) {
        int run = 0;
        for (int e = 0; e < E; ++e) { offs[e] = run; run += counts[e]; }
        offs[E] = run;
    }
}

// ---------------- scatter tokens into per-expert segments ----------------
__global__ void moe_scatter(const int* __restrict__ top_e, const float* __restrict__ top_w,
                            const int* __restrict__ offs, int* __restrict__ cursor,
                            int* __restrict__ perm_tok, float* __restrict__ perm_w,
                            int* __restrict__ pos_of) {
    int t = blockIdx.x * blockDim.x + threadIdx.x;
    if (t >= Tt) return;
#pragma unroll
    for (int k = 0; k < TOPK; ++k) {
        int e   = top_e[t * 2 + k];
        int pos = offs[e] + atomicAdd(&cursor[e], 1);
        perm_tok[pos] = t;
        perm_w[pos]   = top_w[t * 2 + k];
        pos_of[t * 2 + k] = pos;
    }
}

// ---------------- grouped GEMM1: H = gelu(x @ w1[e] + b1[e]) ----------------
// Block: TM=128 routed rows x TN=128 F-cols; K over D in KB=64 double-buffered
// chunks staged by global_load_async_to_lds_b128. 8 waves, each 32x64 (2x4 WMMA).
// w1t is [E][F][D] (K-contiguous) from the transpose-convert pass.
__global__ void __launch_bounds__(256)
moe_gemm1(const bf16* __restrict__ xbf, const bf16* __restrict__ w1t,
          const float* __restrict__ b1, const int* __restrict__ offs,
          const int* __restrict__ perm_tok, bf16* __restrict__ H) {
    const int e    = blockIdx.x / MAX_TILES;
    const int tile = blockIdx.x % MAX_TILES;
    const int seg_start = offs[e];
    const int seg_cnt   = offs[e + 1] - seg_start;
    const int row0 = tile * TM;
    if (row0 >= seg_cnt) return;           // block-uniform: EXEC stays all-ones
    const int n0 = blockIdx.y * TN;

    __shared__ __align__(16) bf16 As[2][TM][KB];
    __shared__ __align__(16) bf16 Bs[2][TN][KB];
    __shared__ int tok[TM];

    const int tid = threadIdx.x;
    for (int i = tid; i < TM; i += 256) {
        int r = min(row0 + i, seg_cnt - 1);     // clamp ragged tail
        tok[i] = perm_tok[seg_start + r];
    }
    __syncthreads();

    auto issue = [&](int buf, int k0) {
#pragma unroll
        for (int i = tid; i < TM * KB / 8; i += 256) {
            int flat = i * 8;
            int r = flat / KB, c = flat % KB;
            async_b128(ldsoff_of(&As[buf][r][c]),
                       xbf + (size_t)tok[r] * D + k0 + c);
        }
#pragma unroll
        for (int i = tid; i < TN * KB / 8; i += 256) {
            int flat = i * 8;
            int n = flat / KB, c = flat % KB;
            async_b128(ldsoff_of(&Bs[buf][n][c]),
                       w1t + ((size_t)e * F + n0 + n) * D + k0 + c);
        }
    };

    const int lane  = tid & 31;
    const int wid   = tid >> 5;
    const int mbase = (wid & 3) * 32;       // 4 waves along M
    const int nbase = (wid >> 2) * 64;      // 2 waves along N
    const int half  = lane >> 4;
    const int l16   = lane & 15;

    v8f acc[2][4] = {};

    issue(0, 0);
    async_wait0();
    __syncthreads();

    int buf = 0;
    for (int k0 = 0; k0 < D; k0 += KB) {
        if (k0 + KB < D) issue(buf ^ 1, k0 + KB);   // prefetch next chunk (DMA)

#pragma unroll
        for (int ks = 0; ks < KB; ks += 32) {
            FragBF a[2];
#pragma unroll
            for (int mi = 0; mi < 2; ++mi) {
                int row = mbase + mi * 16 + l16;
                a[mi].h[0] = *(const v8bf*)&As[buf][row][ks + half * 8];
                a[mi].h[1] = *(const v8bf*)&As[buf][row][ks + 16 + half * 8];
            }
#pragma unroll
            for (int ni = 0; ni < 4; ++ni) {
                int col = nbase + ni * 16 + l16;
                FragBF b;
                b.h[0] = *(const v8bf*)&Bs[buf][col][ks + half * 8];
                b.h[1] = *(const v8bf*)&Bs[buf][col][ks + 16 + half * 8];
#pragma unroll
                for (int mi = 0; mi < 2; ++mi)
                    acc[mi][ni] = __builtin_amdgcn_wmma_f32_16x16x32_bf16(
                        false, a[mi].v, false, b.v, (short)0, acc[mi][ni], false, false);
            }
        }
        async_wait0();      // next chunk landed in LDS
        __syncthreads();    // all waves done reading current buffer
        buf ^= 1;
    }

    // epilogue: bias + exact GELU, store bf16 H
#pragma unroll
    for (int mi = 0; mi < 2; ++mi)
#pragma unroll
        for (int ni = 0; ni < 4; ++ni)
#pragma unroll
            for (int r = 0; r < 8; ++r) {
                int ml = mbase + mi * 16 + half * 8 + r;
                if (row0 + ml < seg_cnt) {
                    int gcol = n0 + nbase + ni * 16 + l16;
                    float v = acc[mi][ni][r] + b1[e * F + gcol];
                    float g = 0.5f * v * (1.0f + erff(v * 0.70710678118654752f));
                    H[(size_t)(seg_start + row0 + ml) * F + gcol] = (bf16)g;
                }
            }
}

// ---------------- grouped GEMM2: partial = w * (H @ w2[e] + b2[e]) ----------------
// w2t is [E][D][F] (K=F contiguous) from the transpose-convert pass.
__global__ void __launch_bounds__(256)
moe_gemm2(const bf16* __restrict__ H, const bf16* __restrict__ w2t,
          const float* __restrict__ b2, const int* __restrict__ offs,
          const float* __restrict__ perm_w, float* __restrict__ partial) {
    const int e    = blockIdx.x / MAX_TILES;
    const int tile = blockIdx.x % MAX_TILES;
    const int seg_start = offs[e];
    const int seg_cnt   = offs[e + 1] - seg_start;
    const int row0 = tile * TM;
    if (row0 >= seg_cnt) return;
    const int n0 = blockIdx.y * TN;

    __shared__ __align__(16) bf16 As[2][TM][KB];
    __shared__ __align__(16) bf16 Bs[2][TN][KB];
    __shared__ int   hrow[TM];
    __shared__ float wrow[TM];

    const int tid = threadIdx.x;
    for (int i = tid; i < TM; i += 256) {
        int r = min(row0 + i, seg_cnt - 1);
        hrow[i] = seg_start + r;
        wrow[i] = perm_w[seg_start + r];
    }
    __syncthreads();

    auto issue = [&](int buf, int k0) {
#pragma unroll
        for (int i = tid; i < TM * KB / 8; i += 256) {
            int flat = i * 8;
            int r = flat / KB, c = flat % KB;
            async_b128(ldsoff_of(&As[buf][r][c]),
                       H + (size_t)hrow[r] * F + k0 + c);
        }
#pragma unroll
        for (int i = tid; i < TN * KB / 8; i += 256) {
            int flat = i * 8;
            int n = flat / KB, c = flat % KB;
            async_b128(ldsoff_of(&Bs[buf][n][c]),
                       w2t + ((size_t)e * D + n0 + n) * F + k0 + c);
        }
    };

    const int lane  = tid & 31;
    const int wid   = tid >> 5;
    const int mbase = (wid & 3) * 32;
    const int nbase = (wid >> 2) * 64;
    const int half  = lane >> 4;
    const int l16   = lane & 15;

    v8f acc[2][4] = {};

    issue(0, 0);
    async_wait0();
    __syncthreads();

    int buf = 0;
    for (int k0 = 0; k0 < F; k0 += KB) {
        if (k0 + KB < F) issue(buf ^ 1, k0 + KB);

#pragma unroll
        for (int ks = 0; ks < KB; ks += 32) {
            FragBF a[2];
#pragma unroll
            for (int mi = 0; mi < 2; ++mi) {
                int row = mbase + mi * 16 + l16;
                a[mi].h[0] = *(const v8bf*)&As[buf][row][ks + half * 8];
                a[mi].h[1] = *(const v8bf*)&As[buf][row][ks + 16 + half * 8];
            }
#pragma unroll
            for (int ni = 0; ni < 4; ++ni) {
                int col = nbase + ni * 16 + l16;
                FragBF b;
                b.h[0] = *(const v8bf*)&Bs[buf][col][ks + half * 8];
                b.h[1] = *(const v8bf*)&Bs[buf][col][ks + 16 + half * 8];
#pragma unroll
                for (int mi = 0; mi < 2; ++mi)
                    acc[mi][ni] = __builtin_amdgcn_wmma_f32_16x16x32_bf16(
                        false, a[mi].v, false, b.v, (short)0, acc[mi][ni], false, false);
            }
        }
        async_wait0();
        __syncthreads();
        buf ^= 1;
    }

#pragma unroll
    for (int mi = 0; mi < 2; ++mi)
#pragma unroll
        for (int ni = 0; ni < 4; ++ni)
#pragma unroll
            for (int r = 0; r < 8; ++r) {
                int ml = mbase + mi * 16 + half * 8 + r;
                if (row0 + ml < seg_cnt) {
                    int gcol = n0 + nbase + ni * 16 + l16;
                    float v = wrow[ml] * (acc[mi][ni][r] + b2[e * D + gcol]);
                    partial[(size_t)(seg_start + row0 + ml) * D + gcol] = v;
                }
            }
}

// ---------------- combine the token's two expert partials ----------------
__global__ void moe_combine(const float* __restrict__ partial, const int* __restrict__ pos_of,
                            float* __restrict__ out) {
    for (int i = blockIdx.x * blockDim.x + threadIdx.x; i < Tt * D;
         i += gridDim.x * blockDim.x) {
        int t = i / D, d = i % D;
        int p0 = pos_of[t * 2 + 0];
        int p1 = pos_of[t * 2 + 1];
        out[i] = partial[(size_t)p0 * D + d] + partial[(size_t)p1 * D + d];
    }
}

// ---------------- launch ----------------
extern "C" void kernel_launch(void* const* d_in, const int* in_sizes, int n_in,
                              void* d_out, int out_size, void* d_ws, size_t ws_size,
                              hipStream_t stream) {
    const float* x      = (const float*)d_in[0];
    const float* gate_w = (const float*)d_in[1];
    const float* w1     = (const float*)d_in[2];
    const float* b1     = (const float*)d_in[3];
    const float* w2     = (const float*)d_in[4];
    const float* b2     = (const float*)d_in[5];
    float* out = (float*)d_out;

    // carve workspace (assumes ws_size >= ~340 MB)
    char* p = (char*)d_ws;
    auto carve = [&](size_t bytes) -> void* {
        void* r = (void*)p;
        p += (bytes + 255) & ~(size_t)255;
        return r;
    };
    bf16*  xbf      = (bf16*) carve((size_t)Tt * D * sizeof(bf16));
    bf16*  w1t      = (bf16*) carve((size_t)E * D * F * sizeof(bf16));   // [E][F][D]
    bf16*  w2t      = (bf16*) carve((size_t)E * F * D * sizeof(bf16));   // [E][D][F]
    bf16*  H        = (bf16*) carve((size_t)TSEL * F * sizeof(bf16));
    float* partial  = (float*)carve((size_t)TSEL * D * sizeof(float));
    float* top_w    = (float*)carve((size_t)Tt * 2 * sizeof(float));
    float* perm_w   = (float*)carve((size_t)TSEL * sizeof(float));
    int*   top_e    = (int*)  carve((size_t)Tt * 2 * sizeof(int));
    int*   perm_tok = (int*)  carve((size_t)TSEL * sizeof(int));
    int*   pos_of   = (int*)  carve((size_t)Tt * 2 * sizeof(int));
    int*   counts   = (int*)  carve(E * sizeof(int));
    int*   offs     = (int*)  carve((E + 1) * sizeof(int));
    int*   cursor   = (int*)  carve(E * sizeof(int));

    moe_init<<<1, 32, 0, stream>>>(counts, cursor);

    moe_cvt_bf16<<<4096, 256, 0, stream>>>(x, xbf, Tt * D);
    {   // w1: [E][D][F] -> w1t [E][F][D]
        dim3 g(F / 32, D / 32, E);
        moe_cvt_tr<<<g, 256, 0, stream>>>(w1, w1t, D, F);
    }
    {   // w2: [E][F][D] -> w2t [E][D][F]
        dim3 g(D / 32, F / 32, E);
        moe_cvt_tr<<<g, 256, 0, stream>>>(w2, w2t, F, D);
    }

    moe_router<<<Tt / 8, 256, 0, stream>>>(x, gate_w, top_w, top_e, counts);
    moe_offsets<<<1, 32, 0, stream>>>(counts, offs);
    moe_scatter<<<Tt / 256, 256, 0, stream>>>(top_e, top_w, offs, cursor,
                                              perm_tok, perm_w, pos_of);

    dim3 g1(E * MAX_TILES, F / TN);
    moe_gemm1<<<g1, 256, 0, stream>>>(xbf, w1t, b1, offs, perm_tok, H);

    dim3 g2(E * MAX_TILES, D / TN);
    moe_gemm2<<<g2, 256, 0, stream>>>(H, w2t, b2, offs, perm_w, partial);

    moe_combine<<<4096, 256, 0, stream>>>(partial, pos_of, out);
}
